// MultiheadAttention_88484916232353
// MI455X (gfx1250) — matrix operand order repaired and checked by reading, not compiled, not run on hardware
//
#include <hip/hip_runtime.h>
#include <hip/hip_bf16.h>

typedef __attribute__((ext_vector_type(16))) _Float16 v16h;
typedef __attribute__((ext_vector_type(8)))  _Float16 v8h;
typedef __attribute__((ext_vector_type(8)))  float    v8f;
typedef __attribute__((ext_vector_type(4)))  unsigned int u32x4;
typedef __attribute__((ext_vector_type(8)))  int      i32x8;
typedef __attribute__((ext_vector_type(4)))  int      i32x4;

constexpr int E  = 1024;
constexpr int H  = 16;
constexpr int S  = 1024;
constexpr int B  = 8;
constexpr int HD = 64;          // E / H
constexpr int M  = S * B;       // 8192 rows for the projection GEMMs

constexpr int SLAB_PAD_H = 40;  // 32 data halfs + 8 pad halfs (80B row, TDM-padded)

// ---------------------------------------------------------------------------
// WMMA fragment helpers (wave32, v_wmma_f32_16x16x32_f16)
// ---------------------------------------------------------------------------
__device__ __forceinline__ v16h make_v16(v8h a, v8h b) {
  union { v16h v; v8h h[2]; } u;
  u.h[0] = a; u.h[1] = b;
  return u.v;
}

// A 16x32 f16 fragment from row-major [.., ld] starting at (row base, k0).
// lane&15 = M row; halves h -> K = k0 + h + 16*hi (16 contiguous halves).
__device__ __forceinline__ v16h frag_a(const _Float16* __restrict__ base, int ld, int k0) {
  const int l  = threadIdx.x & 31;
  const int hi = l >> 4;
  const _Float16* p = base + (size_t)(l & 15) * ld + k0 + 16 * hi;
  return make_v16(*(const v8h*)p, *(const v8h*)(p + 8));
}

// B 32x16 f16 fragment, B[k][n] = src[n][k] (row-major [N,K]) -> GEMM-NT.
// lane&15 = N col; halves h -> K = k0 + h + 16*hi (16 contiguous halves).
__device__ __forceinline__ v16h frag_b_nt(const _Float16* __restrict__ src, int ld, int n0, int k0) {
  const int l  = threadIdx.x & 31;
  const int hi = l >> 4;
  const _Float16* p = src + (size_t)(n0 + (l & 15)) * ld + k0 + 16 * hi;
  return make_v16(*(const v8h*)p, *(const v8h*)(p + 8));
}

// B fragment from a TDM-staged LDS slab: rows = N (64), padded row stride 40 halfs.
__device__ __forceinline__ v16h frag_b_slab(const _Float16* sl, int n0) {
  const int l  = threadIdx.x & 31;
  const int hi = l >> 4;
  const _Float16* p = sl + (n0 + (l & 15)) * SLAB_PAD_H + 16 * hi;
  return make_v16(*(const v8h*)p, *(const v8h*)(p + 8));
}

__device__ __forceinline__ v8f wmma_f16(v16h a, v16h b, v8f c) {
  return __builtin_amdgcn_wmma_f32_16x16x32_f16(false, a, false, b, (short)0, c, false, false);
}

// ---------------------------------------------------------------------------
// Tensor Data Mover: DMA a 64(N-rows) x 32(K-halfs) weight slab into LDS with
// 16B padding every 64B row (-> 80B LDS row stride). 2D D# per CDNA5 ISA 8.3/8.4.
// ---------------------------------------------------------------------------
__device__ __forceinline__ void tdm_load_w_slab(const _Float16* gsrc, _Float16* lds_dst) {
  const uint64_t ga = (uint64_t)(uintptr_t)gsrc;
  const uint32_t la = (uint32_t)(uintptr_t)lds_dst;    // LDS byte offset

  u32x4 g0;
  g0[0] = 1u;                                          // count=1, user descriptor
  g0[1] = la;                                          // lds_addr
  g0[2] = (uint32_t)ga;                                // global_addr[31:0]
  g0[3] = ((uint32_t)(ga >> 32) & 0x01FFFFFFu)         // global_addr[56:32]
          | (2u << 30);                                // type=2 ("image")

  i32x8 g1;
  g1[0] = (1 << 16)                                    // data_size = 2 bytes
        | (1 << 20)                                    // pad_enable
        | (3 << 22)                                    // pad_interval: 16 dwords (=64B row)
        | (3 << 25);                                   // pad_amount: 4 dwords (=16B)
  g1[1] = (int)((1024u & 0xFFFFu) << 16);              // tensor_dim0[15:0]   (K = 1024)
  g1[2] = (int)(((1024u >> 16) & 0xFFFFu)              // tensor_dim0[31:16]
        |        ((1024u & 0xFFFFu) << 16));           // tensor_dim1[15:0]   (N = 1024)
  g1[3] = (int)(((1024u >> 16) & 0xFFFFu)              // tensor_dim1[31:16]
        |        (32u << 16));                         // tile_dim0 = 32 (64B, mult of 4B)
  g1[4] = 64;                                          // tile_dim1 = 64, tile_dim2 = 0
  g1[5] = 1024;                                        // tensor_dim0_stride[31:0] = 1024 elems
  g1[6] = 0;                                           // stride hi / dim1_stride lo
  g1[7] = 0;

  i32x4 z4 = {0, 0, 0, 0};                             // groups 2/3 unused (2D tensor)
#if defined(__clang_major__) && (__clang_major__ >= 23)
  i32x8 z8 = {0, 0, 0, 0, 0, 0, 0, 0};
  __builtin_amdgcn_tensor_load_to_lds(g0, g1, z4, z4, z8, 0);
#else
  __builtin_amdgcn_tensor_load_to_lds(g0, g1, z4, z4, 0);
#endif
}

// ---------------------------------------------------------------------------
// fp32 -> f16 conversion
// ---------------------------------------------------------------------------
__global__ void conv_f32_to_f16(const float* __restrict__ in, _Float16* __restrict__ out, int n) {
  int i = blockIdx.x * blockDim.x + threadIdx.x;
  if (i < n) out[i] = (_Float16)in[i];
}

// ---------------------------------------------------------------------------
// Fused QKV projection: out = query @ W^T (+bias) (*scale) -> head layouts.
// grid = (M/16/4, E/64, 3), block = 128 (4 waves). Weight slab double-buffered
// in LDS via TDM; A fragments and LDS B fragments register double-buffered.
// ---------------------------------------------------------------------------
__global__ void __launch_bounds__(128) qkv_proj_kernel(
    const _Float16* __restrict__ x16,                       // [M, E]
    const _Float16* __restrict__ Wq16,
    const _Float16* __restrict__ Wk16,
    const _Float16* __restrict__ Wv16,                      // [E, E] each
    const float* __restrict__ bq, const float* __restrict__ bv,
    _Float16* __restrict__ qh,                              // [B*H, S, HD]
    _Float16* __restrict__ kh,                              // [B*H, S, HD]
    _Float16* __restrict__ vT)                              // [B*H, HD, S]
{
  __shared__ __attribute__((aligned(16))) _Float16 slab[2][64 * SLAB_PAD_H];

  const int w  = threadIdx.x >> 5;
  const int l  = threadIdx.x & 31;
  const int m0 = (blockIdx.x * 4 + w) * 16;
  const int h  = blockIdx.y;                                // head / 64-wide N block
  const int z  = blockIdx.z;                                // 0=q, 1=k, 2=v

  const _Float16* W16 = (z == 0) ? Wq16 : (z == 1) ? Wk16 : Wv16;
  const _Float16* wbase  = W16 + (size_t)h * 64 * E;        // rows nb..nb+63
  const _Float16* a_base = x16 + (size_t)m0 * E;

  constexpr int NK = E / 32;                                // 32 k-steps

  if (threadIdx.x < 32) tdm_load_w_slab(wbase, slab[0]);    // prologue DMA (wave 0)

  v8f  acc[4] = {};
  v16h a_cur  = frag_a(a_base, E, 0);

  for (int i = 0; i < NK; ++i) {
    if (threadIdx.x < 32) {
      if (i + 1 < NK) {
        tdm_load_w_slab(wbase + (i + 1) * 32, slab[(i + 1) & 1]);
        __builtin_amdgcn_s_wait_tensorcnt(1);               // slab i done, slab i+1 in flight
      } else {
        __builtin_amdgcn_s_wait_tensorcnt(0);               // drain
      }
    }
    __syncthreads();                                        // slab[i&1] ready for all waves

    v16h a_next = a_cur;
    if (i + 1 < NK) a_next = frag_a(a_base, E, (i + 1) * 32);  // prefetch next A

    const _Float16* sl = slab[i & 1];
    v16h bf = frag_b_slab(sl, 0);
#pragma unroll
    for (int j = 0; j < 4; ++j) {
      v16h bnx = bf;
      if (j < 3) bnx = frag_b_slab(sl, (j + 1) * 16);       // LDS prefetch j+1
      acc[j] = wmma_f16(a_cur, bf, acc[j]);
      bf = bnx;
    }
    a_cur = a_next;
    __syncthreads();                                        // protect slab[i&1] from next DMA
  }

  const int   hi    = l >> 4;
  const float scale = (z == 0) ? 0.125f : 1.0f;             // HD^-0.5 = 1/8
#pragma unroll
  for (int j = 0; j < 4; ++j) {
    int   hd   = j * 16 + (l & 15);
    int   n    = h * 64 + hd;
    float bias = (z == 0) ? bq[n] : (z == 2) ? bv[n] : 0.0f;
#pragma unroll
    for (int r = 0; r < 8; ++r) {
      int m = m0 + r + 8 * hi;                              // m = s*B + b
      int s = m >> 3;
      int b = m & 7;
      float val = (acc[j][r] + bias) * scale;
      if (z == 2) {
        vT[((size_t)(b * H + h) * HD + hd) * S + s] = (_Float16)val;
      } else {
        _Float16* dst = (z == 0) ? qh : kh;
        dst[((size_t)(b * H + h) * S + s) * HD + hd] = (_Float16)val;
      }
    }
  }
}

// ---------------------------------------------------------------------------
// Attention: one workgroup (4 waves) per (head, 16-row q tile).
// Phase 1: scores = q@k^T + bias -> f16 in LDS (wave w: k cols [256w,256w+256))
// Phase 2: row softmax (unnormalized; rowsum saved, divide in PV epilogue)
// Phase 3: O = P @ V (wave w -> hd [16w,16w+16)), P A-frags from LDS
// ---------------------------------------------------------------------------
__global__ void __launch_bounds__(128) attention_kernel(
    const _Float16* __restrict__ qh, const _Float16* __restrict__ kh,
    const _Float16* __restrict__ vT, const float* __restrict__ attn_bias,
    float* __restrict__ attn_out)                           // [S, B, E]
{
  __shared__ __attribute__((aligned(16))) _Float16 sc[16][1032];  // padded rows
  __shared__ float redmax[16][8];
  __shared__ float redsum[16][8];
  __shared__ float rowsum[16];

  const int w  = threadIdx.x >> 5;
  const int l  = threadIdx.x & 31;
  const int hi = l >> 4;
  const int s0 = blockIdx.x * 16;
  const int bh = blockIdx.y;

  // ---- Phase 1: scores (K-fragments register double-buffered; bias loads
  //      issued before the WMMAs so the 512MB stream hides under them) ----
  const _Float16* qb = qh + (size_t)bh * S * HD + (size_t)s0 * HD;
  const _Float16* kb = kh + (size_t)bh * S * HD;
  v16h aq0 = frag_a(qb, HD, 0);
  v16h aq1 = frag_a(qb, HD, 32);
  v16h b0  = frag_b_nt(kb, HD, w * 256, 0);
  v16h b1  = frag_b_nt(kb, HD, w * 256, 32);
  for (int t = 0; t < 16; ++t) {
    int  kn0 = w * 256 + t * 16;
    int  n   = kn0 + (l & 15);
    const float* bp = attn_bias + ((size_t)bh * S + s0) * S + n;   // coalesced read
    float bias_r[8];
#pragma unroll
    for (int r = 0; r < 8; ++r) bias_r[r] = bp[(size_t)(r + 8 * hi) * S];

    v16h nb0 = b0, nb1 = b1;
    if (t + 1 < 16) {                                       // prefetch next K tile
      nb0 = frag_b_nt(kb, HD, kn0 + 16, 0);
      nb1 = frag_b_nt(kb, HD, kn0 + 16, 32);
    }
    v8f c = {};
    c = wmma_f16(aq0, b0, c);
    c = wmma_f16(aq1, b1, c);
#pragma unroll
    for (int r = 0; r < 8; ++r) {
      int m = r + 8 * hi;
      sc[m][n] = (_Float16)(c[r] + bias_r[r]);
    }
    b0 = nb0; b1 = nb1;
  }
  __syncthreads();

  // ---- Phase 2: softmax over 1024 cols per row (16 rows x 8 segments) ----
  {
    const int row = threadIdx.x >> 3;
    const int seg = threadIdx.x & 7;
    const int c0  = seg * 128;
    float mx = -3.0e38f;
    for (int c = 0; c < 128; ++c) mx = fmaxf(mx, (float)sc[row][c0 + c]);
    redmax[row][seg] = mx;
    __syncthreads();
    float rmax = redmax[row][0];
#pragma unroll
    for (int k = 1; k < 8; ++k) rmax = fmaxf(rmax, redmax[row][k]);
    float sum = 0.0f;
    for (int c = 0; c < 128; ++c) {
      float p = __expf((float)sc[row][c0 + c] - rmax);
      sc[row][c0 + c] = (_Float16)p;
      sum += p;
    }
    redsum[row][seg] = sum;
    __syncthreads();
    if (seg == 0) {
      float tsum = 0.0f;
#pragma unroll
      for (int k = 0; k < 8; ++k) tsum += redsum[row][k];
      rowsum[row] = tsum;
    }
    __syncthreads();
  }

  // ---- Phase 3: O = P @ V (P frags from LDS + V frags from global,
  //      both register double-buffered one k-step ahead) ----
  const int hd0 = w * 16;
  v8f acc = {};
  const _Float16* vb = vT + ((size_t)bh * HD + hd0 + (l & 15)) * S;

  const _Float16* ap = &sc[l & 15][16 * hi];
  v16h a = make_v16(*(const v8h*)ap, *(const v8h*)(ap + 8));
  const _Float16* bp = vb + 16 * hi;
  v16h b = make_v16(*(const v8h*)bp, *(const v8h*)(bp + 8));

  for (int kc = 0; kc < S; kc += 32) {
    v16h an = a, bn2 = b;
    if (kc + 32 < S) {
      const _Float16* apn = &sc[l & 15][kc + 32 + 16 * hi];
      an  = make_v16(*(const v8h*)apn, *(const v8h*)(apn + 8));
      const _Float16* bpn = vb + kc + 32 + 16 * hi;
      bn2 = make_v16(*(const v8h*)bpn, *(const v8h*)(bpn + 8));
    }
    acc = wmma_f16(a, b, acc);
    a = an; b = bn2;
  }
  const int bb = bh / H, hh = bh % H;
#pragma unroll
  for (int r = 0; r < 8; ++r) {
    int   m   = r + 8 * hi;
    float val = acc[r] / rowsum[m];
    attn_out[((size_t)(s0 + m) * B + bb) * E + hh * HD + hd0 + (l & 15)] = val;
  }
}

// ---------------------------------------------------------------------------
// LayerNorm over E=1024, one block (256 thr) per row; f16 output for out-proj
// ---------------------------------------------------------------------------
__global__ void __launch_bounds__(256) layernorm_kernel(
    const float* __restrict__ x, const float* __restrict__ g,
    const float* __restrict__ bn, _Float16* __restrict__ y)
{
  __shared__ float rs[256];
  __shared__ float rq[256];
  const int row = blockIdx.x;
  const int t   = threadIdx.x;
  const float* xr = x + (size_t)row * E;
  float4 xv = ((const float4*)xr)[t];
  rs[t] = xv.x + xv.y + xv.z + xv.w;
  rq[t] = xv.x * xv.x + xv.y * xv.y + xv.z * xv.z + xv.w * xv.w;
  __syncthreads();
  for (int o = 128; o > 0; o >>= 1) {
    if (t < o) { rs[t] += rs[t + o]; rq[t] += rq[t + o]; }
    __syncthreads();
  }
  const float mu  = rs[0] * (1.0f / E);
  const float var = rq[0] * (1.0f / E) - mu * mu;
  const float inv = rsqrtf(var + 1e-5f);
  const float vx[4] = {xv.x, xv.y, xv.z, xv.w};
#pragma unroll
  for (int i = 0; i < 4; ++i) {
    int e = t * 4 + i;
    y[(size_t)row * E + e] = (_Float16)((vx[i] - mu) * inv * g[e] + bn[e]);
  }
}

// ---------------------------------------------------------------------------
// Output projection: out = ln16 @ Wo^T + bo, fp32 result. Same TDM-staged
// pipeline as QKV. grid = (M/16/4, E/64), block = 128.
// ---------------------------------------------------------------------------
__global__ void __launch_bounds__(128) out_proj_kernel(
    const _Float16* __restrict__ x16, const _Float16* __restrict__ Wo16,
    const float* __restrict__ bo, float* __restrict__ out)
{
  __shared__ __attribute__((aligned(16))) _Float16 slab[2][64 * SLAB_PAD_H];

  const int w  = threadIdx.x >> 5;
  const int l  = threadIdx.x & 31;
  const int m0 = (blockIdx.x * 4 + w) * 16;
  const int nb = blockIdx.y * 64;
  const _Float16* wbase  = Wo16 + (size_t)nb * E;
  const _Float16* a_base = x16 + (size_t)m0 * E;

  constexpr int NK = E / 32;

  if (threadIdx.x < 32) tdm_load_w_slab(wbase, slab[0]);

  v8f  acc[4] = {};
  v16h a_cur  = frag_a(a_base, E, 0);

  for (int i = 0; i < NK; ++i) {
    if (threadIdx.x < 32) {
      if (i + 1 < NK) {
        tdm_load_w_slab(wbase + (i + 1) * 32, slab[(i + 1) & 1]);
        __builtin_amdgcn_s_wait_tensorcnt(1);
      } else {
        __builtin_amdgcn_s_wait_tensorcnt(0);
      }
    }
    __syncthreads();

    v16h a_next = a_cur;
    if (i + 1 < NK) a_next = frag_a(a_base, E, (i + 1) * 32);

    const _Float16* sl = slab[i & 1];
    v16h bf = frag_b_slab(sl, 0);
#pragma unroll
    for (int j = 0; j < 4; ++j) {
      v16h bnx = bf;
      if (j < 3) bnx = frag_b_slab(sl, (j + 1) * 16);
      acc[j] = wmma_f16(a_cur, bf, acc[j]);
      bf = bnx;
    }
    a_cur = a_next;
    __syncthreads();
  }

  const int hi = l >> 4;
#pragma unroll
  for (int j = 0; j < 4; ++j) {
    int   n    = nb + j * 16 + (l & 15);
    float bias = bo[n];
#pragma unroll
    for (int r = 0; r < 8; ++r) {
      int m = m0 + r + 8 * hi;
      out[(size_t)m * E + n] = acc[j][r] + bias;
    }
  }
}

// ---------------------------------------------------------------------------
// Host launcher
// ---------------------------------------------------------------------------
extern "C" void kernel_launch(void* const* d_in, const int* in_sizes, int n_in,
                              void* d_out, int out_size, void* d_ws, size_t ws_size,
                              hipStream_t stream) {
  (void)in_sizes; (void)n_in; (void)out_size; (void)ws_size;

  const float* query     = (const float*)d_in[0];
  const float* attn_bias = (const float*)d_in[1];
  const float* Wq        = (const float*)d_in[2];
  const float* bq        = (const float*)d_in[3];
  const float* Wk        = (const float*)d_in[4];
  const float* Wv        = (const float*)d_in[5];
  const float* bv        = (const float*)d_in[6];
  const float* Wo        = (const float*)d_in[7];
  const float* bo        = (const float*)d_in[8];
  const float* ln_g      = (const float*)d_in[9];
  const float* ln_b      = (const float*)d_in[10];
  float* out = (float*)d_out;

  char* ws = (char*)d_ws;
  _Float16* x16  = (_Float16*)(ws);                           // 16 MB   [M,E] f16
  _Float16* Wq16 = (_Float16*)(ws + ((size_t)16 << 20));      //  2 MB
  _Float16* Wk16 = (_Float16*)(ws + ((size_t)18 << 20));      //  2 MB
  _Float16* Wv16 = (_Float16*)(ws + ((size_t)20 << 20));      //  2 MB
  _Float16* Wo16 = (_Float16*)(ws + ((size_t)22 << 20));      //  2 MB
  _Float16* qh   = (_Float16*)(ws + ((size_t)24 << 20));      // 16 MB   [BH,S,HD]
  _Float16* kh   = (_Float16*)(ws + ((size_t)40 << 20));      // 16 MB   [BH,S,HD]
  _Float16* vT   = (_Float16*)(ws + ((size_t)56 << 20));      // 16 MB   [BH,HD,S]
  float*    attn = (float*)   (ws + ((size_t)72 << 20));      // 32 MB   [S,B,E] f32
  _Float16* ln16 = (_Float16*)(ws + ((size_t)104 << 20));     // 16 MB   [M,E] f16

  conv_f32_to_f16<<<(M * E + 255) / 256, 256, 0, stream>>>(query, x16, M * E);
  conv_f32_to_f16<<<(E * E + 255) / 256, 256, 0, stream>>>(Wq, Wq16, E * E);
  conv_f32_to_f16<<<(E * E + 255) / 256, 256, 0, stream>>>(Wk, Wk16, E * E);
  conv_f32_to_f16<<<(E * E + 255) / 256, 256, 0, stream>>>(Wv, Wv16, E * E);
  conv_f32_to_f16<<<(E * E + 255) / 256, 256, 0, stream>>>(Wo, Wo16, E * E);

  qkv_proj_kernel<<<dim3(M / 16 / 4, E / 64, 3), 128, 0, stream>>>(
      x16, Wq16, Wk16, Wv16, bq, bv, qh, kh, vT);

  attention_kernel<<<dim3(S / 16, B * H), 128, 0, stream>>>(
      qh, kh, vT, attn_bias, attn);

  layernorm_kernel<<<M, 256, 0, stream>>>(attn, ln_g, ln_b, ln16);

  out_proj_kernel<<<dim3(M / 16 / 4, E / 64), 128, 0, stream>>>(ln16, Wo16, bo, out);
}
